// MoeSpaceTimeModel_80882824118314
// MI455X (gfx1250) — matrix operand order repaired and checked by reading, not compiled, not run on hardware
//
#include <hip/hip_runtime.h>
#include <math.h>

// ---------------------------------------------------------------------------
// MoeSpaceTimeModel for MI455X (gfx1250, wave32).
// 4-kernel pipeline; the heavy render-expert MLPs (8 experts x 3x [64x128]
// GEMMs over 131072 points ~= 51.5 GFLOP) run on v_wmma_f32_16x16x32_f16 with
// per-workgroup LDS-staged f16 transposed weights.
// Transcendentals use the HW paths (v_sin/v_cos/v_exp) to keep the render
// stage matrix-bound instead of libm-argument-reduction-bound.
//
// Workspace layout (d_ws): flat idx (N i32) | feats (N*16 f32) |
//   wvec (N*8 f32) | routs (8*N*8 f32)  ~= 46 MB total.
// Input pointer order = setup_inputs() dict insertion order, recursively.
// ---------------------------------------------------------------------------

typedef _Float16 h16;
typedef __attribute__((ext_vector_type(16))) _Float16 v16h;
typedef __attribute__((ext_vector_type(8)))  _Float16 v8h;
typedef __attribute__((ext_vector_type(8)))  float    v8f;

#define DEV __device__ __forceinline__

// ---- fast transcendentals (hardware v_exp/v_sin/v_cos) ----
DEV float fexpf_(float x){ return __expf(x); }
DEV float fsinf_(float x){ return __sinf(x); }
DEV float fcosf_(float x){ return __cosf(x); }
DEV float siluf(float x){ return x / (1.f + fexpf_(-x)); }
// tanh(x) = 1 - 2/(e^{2x}+1): e->inf => 1, e->0 => -1 (no NaN at extremes)
DEV float tanhf_(float x){ return 1.f - 2.f/(fexpf_(2.f*x) + 1.f); }
DEV float geluf(float x){
  return 0.5f*x*(1.f + tanhf_(0.7978845608028654f*(x + 0.044715f*x*x*x)));
}
DEV float sigmf(float x){ return 1.f/(1.f+fexpf_(-x)); }

// top-2 softmax scatter over 8 logits (jax.lax.top_k + softmax + one_hot sum)
DEV void top2w8(const float* lg, float* w){
  int i1=0; float m1=lg[0];
  #pragma unroll
  for (int e=1;e<8;e++) if (lg[e]>m1){ m1=lg[e]; i1=e; }
  int i2=0; float m2=-3.4e38f;
  #pragma unroll
  for (int e=0;e<8;e++) if (e!=i1 && lg[e]>m2){ m2=lg[e]; i2=e; }
  float p1 = 1.f/(1.f+fexpf_(m2-m1));
  float p2 = 1.f-p1;
  #pragma unroll
  for (int e=0;e<8;e++) w[e] = (e==i1)?p1:((e==i2)?p2:0.f);
}

// ============================ Kernel 1: geo =================================
struct GeoParams {
  const float *ge_proj,*ge_b1,*ge_b2,*ge_win,*ge_rms0,*ge_rms1,*ge_w0,*ge_w1,*ge_wf,*ge_bf;
  const float *gg_proj,*gg_b1,*gg_b2,*gg_win,*gg_rms0,*gg_w0,*gg_wf,*gg_bf;
};

__global__ void __launch_bounds__(256)
k_geo(const float* __restrict__ pt, const float* __restrict__ tm,
      GeoParams P, int* __restrict__ flat, int n)
{
  int i = blockIdx.x*256 + threadIdx.x;
  if (i >= n) return;
  float px=pt[3*i], py=pt[3*i+1], pz=pt[3*i+2], t=tm[i];
  float rho   = sqrtf(px*px+py*py+pz*pz);
  float phi   = atan2f(py, px);
  float cth   = fminf(1.f, fmaxf(-1.f, pz/rho));
  float theta = acosf(cth);

  // ---- gate net: 12 -> geglu16->8 -> geglu16->8 -> 8 logits ----
  float gl[8];
  {
    float emb[12];
    #pragma unroll
    for (int c=0;c<8;c++)
      emb[c] = rho*P.gg_proj[c] + phi*P.gg_proj[8+c] + theta*P.gg_proj[16+c];
    float c0 = fcosf_(t + P.gg_b1[0]);
    float s0 = fsinf_(t + P.gg_b2[0]);
    emb[8]=c0; emb[9]=s0; emb[10]=siluf(c0); emb[11]=siluf(s0);
    float h[8];
    #pragma unroll
    for (int j=0;j<8;j++){
      float a=0.f,b=0.f;
      #pragma unroll
      for (int k=0;k<12;k++){ a+=emb[k]*P.gg_win[k*16+j]; b+=emb[k]*P.gg_win[k*16+8+j]; }
      h[j]=a*geluf(b);
    }
    float ss=0.f;
    #pragma unroll
    for (int j=0;j<8;j++) ss+=h[j]*h[j];
    float inv = 1.f/(sqrtf(ss)*0.35355339059327373f + 1e-8f);  // d=8
    float hn[8];
    #pragma unroll
    for (int j=0;j<8;j++) hn[j]=P.gg_rms0[j]*h[j]*inv;
    float h2[8];
    #pragma unroll
    for (int j=0;j<8;j++){
      float a=0.f,b=0.f;
      #pragma unroll
      for (int k=0;k<8;k++){ a+=hn[k]*P.gg_w0[k*16+j]; b+=hn[k]*P.gg_w0[k*16+8+j]; }
      h2[j]=a*geluf(b);
    }
    #pragma unroll
    for (int j=0;j<8;j++){
      float o=P.gg_bf[j];
      #pragma unroll
      for (int k=0;k<8;k++) o+=h2[k]*P.gg_wf[k*8+j];
      gl[j]=o;
    }
  }
  float gw[8]; top2w8(gl, gw);

  // ---- 8 geo experts, dense (matches reference vmap) ----
  float geo[4]={0.f,0.f,0.f,0.f};
  for (int e=0;e<8;e++){
    float we = (e==0)?1.f:gw[e-1];     // gw[:, :7] weights experts 1..7
    float emb[12];
    const float* pr = P.ge_proj + e*24;
    #pragma unroll
    for (int c=0;c<8;c++)
      emb[c] = rho*pr[c] + phi*pr[8+c] + theta*pr[16+c];
    float c0 = fcosf_(t + P.ge_b1[e]);
    float s0 = fsinf_(t + P.ge_b2[e]);
    emb[8]=c0; emb[9]=s0; emb[10]=siluf(c0); emb[11]=siluf(s0);
    float h[16];
    const float* win = P.ge_win + e*12*32;
    #pragma unroll
    for (int j=0;j<16;j++){
      float a=0.f,b=0.f;
      #pragma unroll
      for (int k=0;k<12;k++){ a+=emb[k]*win[k*32+j]; b+=emb[k]*win[k*32+16+j]; }
      h[j]=a*geluf(b);
    }
    #pragma unroll
    for (int L=0;L<2;L++){
      const float* rs = (L==0?P.ge_rms0:P.ge_rms1) + e*16;
      const float* w  = (L==0?P.ge_w0 :P.ge_w1 ) + e*16*32;
      float ss=0.f;
      #pragma unroll
      for (int j=0;j<16;j++) ss+=h[j]*h[j];
      float inv = 1.f/(sqrtf(ss)*0.25f + 1e-8f);               // d=16
      float hn[16];
      #pragma unroll
      for (int j=0;j<16;j++) hn[j]=rs[j]*h[j]*inv;
      float h2[16];
      #pragma unroll
      for (int j=0;j<16;j++){
        float a=0.f,b=0.f;
        #pragma unroll
        for (int k=0;k<16;k++){ a+=hn[k]*w[k*32+j]; b+=hn[k]*w[k*32+16+j]; }
        h2[j]=a*geluf(b);
      }
      #pragma unroll
      for (int j=0;j<16;j++) h[j]=h2[j];
    }
    const float* wf = P.ge_wf + e*16*4;
    #pragma unroll
    for (int d=0;d<4;d++){
      float o = P.ge_bf[e*4+d];
      #pragma unroll
      for (int k=0;k<16;k++) o+=h[k]*wf[k*4+d];
      geo[d]+=we*o;
    }
  }
  int f=0;
  #pragma unroll
  for (int d=0;d<4;d++){
    float u = sigmf(geo[d]);
    int id = (int)(u*31.f);
    id = id<0?0:(id>31?31:id);
    f = f*32 + id;
  }
  flat[i]=f;
}

// =================== Kernel 2: table gate + gather + feats ==================
struct TableParams {
  const float *cb,*sb,*win,*bin,*rms0,*w0,*b0,*wf,*bf;
  const float *tables,*rgate;
};

__global__ void __launch_bounds__(256)
k_table(const float* __restrict__ pt, const float* __restrict__ tm,
        const int* __restrict__ flat, TableParams P,
        float* __restrict__ feats, float* __restrict__ wvec, int n)
{
  int i = blockIdx.x*256 + threadIdx.x;
  if (i >= n) return;
  float x[4]={pt[3*i],pt[3*i+1],pt[3*i+2],tm[i]};
  float sb=P.sb[0], cb=P.cb[0];
  float e[16];
  #pragma unroll
  for (int d=0;d<4;d++){
    float xs=x[d]+sb, xc=x[d]+cb;
    e[d]=fsinf_(xs); e[4+d]=fsinf_(siluf(xs));
    e[8+d]=fcosf_(xc); e[12+d]=fcosf_(siluf(xc));
  }
  float h[8];
  #pragma unroll
  for (int j=0;j<8;j++){
    float a=P.bin[j], b=P.bin[8+j];
    #pragma unroll
    for (int k=0;k<16;k++){ a+=e[k]*P.win[k*16+j]; b+=e[k]*P.win[k*16+8+j]; }
    h[j]=a*geluf(b);
  }
  float ss=0.f;
  #pragma unroll
  for (int j=0;j<8;j++) ss+=h[j]*h[j];
  float inv = 1.f/(sqrtf(ss)*0.35355339059327373f + 1e-8f);
  float hn[8];
  #pragma unroll
  for (int j=0;j<8;j++) hn[j]=P.rms0[j]*h[j]*inv;
  float h2[8];
  #pragma unroll
  for (int j=0;j<8;j++){
    float a=P.b0[j], b=P.b0[8+j];
    #pragma unroll
    for (int k=0;k<8;k++){ a+=hn[k]*P.w0[k*16+j]; b+=hn[k]*P.w0[k*16+8+j]; }
    h2[j]=a*geluf(b);
  }
  float lg[6];
  #pragma unroll
  for (int j=0;j<6;j++){
    float o=P.bf[j];
    #pragma unroll
    for (int k=0;k<8;k++) o+=h2[k]*P.wf[k*6+j];
    lg[j]=o;
  }
  float mx=lg[0];
  #pragma unroll
  for (int j=1;j<6;j++) mx=fmaxf(mx,lg[j]);
  float tp[6], s=0.f;
  #pragma unroll
  for (int j=0;j<6;j++){ tp[j]=fexpf_(lg[j]-mx); s+=tp[j]; }
  float is=1.f/s;
  #pragma unroll
  for (int j=0;j<6;j++) tp[j]*=is;

  // gather 6 tables (each 32^4 x 16 f32) and mix
  int fi = flat[i];
  const size_t TS = (size_t)1048576*16;
  const float* tb = P.tables + (size_t)fi*16;
  float ft[16];
  #pragma unroll
  for (int k=0;k<16;k++){
    float v = tb[k] + tb[TS+k]
            + tp[0]*tb[2*TS+k] + tp[1]*tb[3*TS+k]
            + tp[2]*tb[4*TS+k] + tp[3]*tb[5*TS+k];
    ft[k]=v;
    feats[(size_t)i*16+k]=v;
  }
  // render gate: feats @ (16x8), top-2 softmax; wvec = [1, rw[:7]]
  float rl[8];
  #pragma unroll
  for (int j=0;j<8;j++){
    float o=0.f;
    #pragma unroll
    for (int k=0;k<16;k++) o+=ft[k]*P.rgate[k*8+j];
    rl[j]=o;
  }
  float rw[8]; top2w8(rl, rw);
  wvec[(size_t)i*8+0]=1.f;
  #pragma unroll
  for (int r=1;r<8;r++) wvec[(size_t)i*8+r]=rw[r-1];
}

// ================= Kernel 3: render experts (WMMA f16) ======================
struct RenderParams {
  const float *cb,*sb,*win,*bin,*rms0,*rms1,*w0,*w1,*b0,*b1,*wf,*bf;
};

#define WST 72   // padded LDS row stride in halfs (144 B = 36 banks, conflict-free)

// geglu on accumulators (lane-local: n-tiles 0..3 = a, 4..7 = b), optional
// rmsnorm over the 64-wide row (shfl_xor reduction across 16 lanes), then
// store f16 tile row-major to this wave's private LDS region.
DEV void geglu_store(const v8f* acc, const float* rmsScale, h16* tile,
                     int col, int hs, bool norm)
{
  float g[8][4]; float ssq[8];
  #pragma unroll
  for (int q=0;q<8;q++){
    ssq[q]=0.f;
    #pragma unroll
    for (int nt=0;nt<4;nt++){
      float v = acc[nt][q]*geluf(acc[nt+4][q]);
      g[q][nt]=v; ssq[q]+=v*v;
    }
  }
  #pragma unroll
  for (int q=0;q<8;q++){
    float scale = 1.f;
    if (norm){
      float s = ssq[q];
      s += __shfl_xor(s,1); s += __shfl_xor(s,2);
      s += __shfl_xor(s,4); s += __shfl_xor(s,8);
      scale = 1.f/(sqrtf(s)*0.125f + 1e-8f);                  // d=64
    }
    int m = q + 8*hs;
    #pragma unroll
    for (int nt=0;nt<4;nt++){
      int c = nt*16+col;
      float v = g[q][nt]*scale;
      if (norm) v *= rmsScale[c];
      tile[m*WST+c] = (h16)v;
    }
  }
  // wave-private LDS RAW fence before fragment reloads
  asm volatile("s_wait_dscnt 0x0" ::: "memory");
}

// (16x64) @ (64x128) with A from LDS h-tile and B from LDS W^T, bias init.
DEV void gemm_tile(v8f* acc, const h16* tile, const h16* wT,
                   const float* bias, int col, int hs)
{
  #pragma unroll
  for (int nt=0;nt<8;nt++){
    float bv = bias[nt*16+col];
    #pragma unroll
    for (int q=0;q<8;q++) acc[nt][q]=bv;
  }
  #pragma unroll
  for (int kc=0;kc<2;kc++){
    // A frag: lane row = col; 16 contiguous halfs k = kc*32 + hs*16
    const h16* ap = tile + col*WST + kc*32 + hs*16;
    v8h alo = *(const v8h*)ap;
    v8h ahi = *(const v8h*)(ap+8);
    v16h A;
    #pragma unroll
    for (int q=0;q<8;q++){ A[q]=alo[q]; A[8+q]=ahi[q]; }
    #pragma unroll
    for (int nt=0;nt<8;nt++){
      // B frag: lane column = col; 16 contiguous halfs k = kc*32 + hs*16
      const h16* bp = wT + (nt*16+col)*WST + kc*32 + hs*16;
      v8h blo = *(const v8h*)bp;
      v8h bhi = *(const v8h*)(bp+8);
      v16h B;
      #pragma unroll
      for (int q=0;q<8;q++){ B[q]=blo[q]; B[8+q]=bhi[q]; }
      acc[nt] = __builtin_amdgcn_wmma_f32_16x16x32_f16(
          false, A, false, B, (short)0, acc[nt], false, false);
    }
  }
}

__global__ void __launch_bounds__(256)
k_render(const float* __restrict__ feats, RenderParams P,
         float* __restrict__ routs, int n, int nblk)
{
  __shared__ alignas(16) h16 sWin[128*WST];
  __shared__ alignas(16) h16 sW0 [128*WST];
  __shared__ alignas(16) h16 sW1 [128*WST];
  __shared__ alignas(16) h16 sWf [16*WST];
  __shared__ alignas(16) h16 sTile[8][16*WST];
  __shared__ float sBin[128], sB0[128], sB1[128], sRms0[64], sRms1[64], sBf[8];

  int r   = blockIdx.x / nblk;       // expert
  int blk = blockIdx.x % nblk;       // 1024-point block
  int tid = threadIdx.x;

  // ---- stage expert weights: f32 global -> f16 transposed LDS ----
  const float* wing = P.win + (size_t)r*64*128;
  const float* w0g  = P.w0  + (size_t)r*64*128;
  const float* w1g  = P.w1  + (size_t)r*64*128;
  const float* wfg  = P.wf  + (size_t)r*64*8;
  for (int idx=tid; idx<128*64; idx+=256){
    int nn = idx>>6, kk = idx&63;
    sWin[nn*WST+kk] = (h16)wing[kk*128+nn];
    sW0 [nn*WST+kk] = (h16)w0g [kk*128+nn];
    sW1 [nn*WST+kk] = (h16)w1g [kk*128+nn];
  }
  for (int idx=tid; idx<16*64; idx+=256){
    int nn=idx>>6, kk=idx&63;
    sWf[nn*WST+kk] = (nn<8)?(h16)wfg[kk*8+nn]:(h16)0.f;   // pad N 8->16
  }
  if (tid<128){ sBin[tid]=P.bin[r*128+tid]; sB0[tid]=P.b0[r*128+tid]; sB1[tid]=P.b1[r*128+tid]; }
  if (tid<64){ sRms0[tid]=P.rms0[r*64+tid]; sRms1[tid]=P.rms1[r*64+tid]; }
  if (tid<8)  sBf[tid]=P.bf[r*8+tid];
  __syncthreads();

  float cb=P.cb[r], sb=P.sb[r];
  int wave = tid>>5, lane = tid&31;
  int col = lane&15, hs = lane>>4;
  h16* tile = &sTile[wave][0];

  for (int tIt=0; tIt<8; ++tIt){                 // 8 tiles x 16 pts per wave
    int base = blk*1024 + wave*128 + tIt*16;
    if (base >= n) break;                        // uniform per wave
    int row = base + col; if (row >= n) row = n-1;
    float f[16];
    const float* fp = feats + (size_t)row*16;
    #pragma unroll
    for (int k=0;k<16;k++) f[k]=fp[k];

    // ---- layer 1: A = e(feats) computed in-register, B = sWin ----
    v8f acc[8];
    #pragma unroll
    for (int nt=0;nt<8;nt++){
      float bv = sBin[nt*16+col];
      #pragma unroll
      for (int q=0;q<8;q++) acc[nt][q]=bv;
    }
    #pragma unroll
    for (int kc=0;kc<2;kc++){
      v16h A;
      #pragma unroll
      for (int j=0;j<8;j++){
        #pragma unroll
        for (int tt=0;tt<2;tt++){
          int k  = kc*32 + hs*16 + 2*j + tt;  // contiguous k = 16*hs + s
          int kk = k & 15, grp = k >> 4;
          float xv;
          if      (grp==0) xv = fsinf_(f[kk]+sb);
          else if (grp==1) xv = fsinf_(siluf(f[kk]+sb));
          else if (grp==2) xv = fcosf_(f[kk]+cb);
          else             xv = fcosf_(siluf(f[kk]+cb));
          A[2*j+tt] = (h16)xv;
        }
      }
      #pragma unroll
      for (int nt=0;nt<8;nt++){
        const h16* bp = &sWin[(nt*16+col)*WST + kc*32 + hs*16];
        v8h blo = *(const v8h*)bp;
        v8h bhi = *(const v8h*)(bp+8);
        v16h B;
        #pragma unroll
        for (int q=0;q<8;q++){ B[q]=blo[q]; B[8+q]=bhi[q]; }
        acc[nt] = __builtin_amdgcn_wmma_f32_16x16x32_f16(
            false, A, false, B, (short)0, acc[nt], false, false);
      }
    }
    geglu_store(acc, sRms0, tile, col, hs, true);
    gemm_tile(acc, tile, sW0, sB0, col, hs);         // layer 2
    geglu_store(acc, sRms1, tile, col, hs, true);
    gemm_tile(acc, tile, sW1, sB1, col, hs);         // layer 3
    geglu_store(acc, nullptr, tile, col, hs, false); // no norm before wf

    // ---- final (16x64)@(64x8), N padded to 16 ----
    v8f accf;
    float bfv = (col<8)?sBf[col]:0.f;
    #pragma unroll
    for (int q=0;q<8;q++) accf[q]=bfv;
    #pragma unroll
    for (int kc=0;kc<2;kc++){
      const h16* ap = tile + col*WST + kc*32 + hs*16;
      v8h alo = *(const v8h*)ap;
      v8h ahi = *(const v8h*)(ap+8);
      v16h A;
      #pragma unroll
      for (int q=0;q<8;q++){ A[q]=alo[q]; A[8+q]=ahi[q]; }
      const h16* bp = &sWf[col*WST + kc*32 + hs*16];
      v8h blo = *(const v8h*)bp;
      v8h bhi = *(const v8h*)(bp+8);
      v16h B;
      #pragma unroll
      for (int q=0;q<8;q++){ B[q]=blo[q]; B[8+q]=bhi[q]; }
      accf = __builtin_amdgcn_wmma_f32_16x16x32_f16(
          false, A, false, B, (short)0, accf, false, false);
    }
    // D layout: lane holds col=lane&15, rows q+8*hs
    if (col < 8){
      #pragma unroll
      for (int q=0;q<8;q++){
        int pr2 = base + q + 8*hs;
        if (pr2 < n)
          routs[((size_t)r*n + pr2)*8 + col] = accf[q];
      }
    }
  }
}

// ======================= Kernel 4: combine + project ========================
__global__ void __launch_bounds__(256)
k_final(const float* __restrict__ routs, const float* __restrict__ wvec,
        const float* __restrict__ origin,
        const float* __restrict__ aw, const float* __restrict__ ab,
        const float* __restrict__ cw, const float* __restrict__ cbv,
        float* __restrict__ out, int n)
{
  int i = blockIdx.x*256 + threadIdx.x;
  if (i >= n) return;
  float alpha = ab[0];
  float c0=cbv[0], c1=cbv[1], c2=cbv[2];
  #pragma unroll
  for (int rr=0;rr<8;rr++){
    float w = wvec[(size_t)i*8+rr];
    const float* rp = routs + ((size_t)rr*n + i)*8;
    #pragma unroll
    for (int j=0;j<8;j++){
      float v = rp[j]*w;
      int k = rr*8+j;
      alpha += v*aw[k];
      c0 += v*cw[k*3+0]; c1 += v*cw[k*3+1]; c2 += v*cw[k*3+2];
    }
  }
  float ox=origin[3*i], oy=origin[3*i+1], oz=origin[3*i+2];
  c0 += ox*cw[64*3+0] + oy*cw[65*3+0] + oz*cw[66*3+0] + alpha*cw[67*3+0];
  c1 += ox*cw[64*3+1] + oy*cw[65*3+1] + oz*cw[66*3+1] + alpha*cw[67*3+1];
  c2 += ox*cw[64*3+2] + oy*cw[65*3+2] + oz*cw[66*3+2] + alpha*cw[67*3+2];
  out[(size_t)i*4+0]=c0; out[(size_t)i*4+1]=c1;
  out[(size_t)i*4+2]=c2; out[(size_t)i*4+3]=alpha;
}

// ============================== launch ======================================
extern "C" void kernel_launch(void* const* d_in, const int* in_sizes, int n_in,
                              void* d_out, int out_size, void* d_ws, size_t ws_size,
                              hipStream_t stream)
{
  (void)n_in; (void)out_size; (void)ws_size;
  const float* point  = (const float*)d_in[0];
  const float* origin = (const float*)d_in[1];
  const float* timev  = (const float*)d_in[2];
  int n = in_sizes[2];

  GeoParams GP;
  GP.ge_proj=(const float*)d_in[3];  GP.ge_b1 =(const float*)d_in[4];
  GP.ge_b2 =(const float*)d_in[5];   GP.ge_win=(const float*)d_in[6];
  GP.ge_rms0=(const float*)d_in[7];  GP.ge_rms1=(const float*)d_in[8];
  GP.ge_w0 =(const float*)d_in[9];   GP.ge_w1 =(const float*)d_in[10];
  GP.ge_wf =(const float*)d_in[11];  GP.ge_bf =(const float*)d_in[12];
  GP.gg_proj=(const float*)d_in[13]; GP.gg_b1 =(const float*)d_in[14];
  GP.gg_b2 =(const float*)d_in[15];  GP.gg_win=(const float*)d_in[16];
  GP.gg_rms0=(const float*)d_in[17]; GP.gg_w0 =(const float*)d_in[18];
  GP.gg_wf =(const float*)d_in[19];  GP.gg_bf =(const float*)d_in[20];

  TableParams TP;
  TP.cb  =(const float*)d_in[21]; TP.sb  =(const float*)d_in[22];
  TP.win =(const float*)d_in[23]; TP.bin =(const float*)d_in[24];
  TP.rms0=(const float*)d_in[25]; TP.w0  =(const float*)d_in[26];
  TP.b0  =(const float*)d_in[27]; TP.wf  =(const float*)d_in[28];
  TP.bf  =(const float*)d_in[29]; TP.tables=(const float*)d_in[30];
  TP.rgate=(const float*)d_in[43];

  RenderParams RP;
  RP.cb  =(const float*)d_in[31]; RP.sb  =(const float*)d_in[32];
  RP.win =(const float*)d_in[33]; RP.bin =(const float*)d_in[34];
  RP.rms0=(const float*)d_in[35]; RP.rms1=(const float*)d_in[36];
  RP.w0  =(const float*)d_in[37]; RP.w1  =(const float*)d_in[38];
  RP.b0  =(const float*)d_in[39]; RP.b1  =(const float*)d_in[40];
  RP.wf  =(const float*)d_in[41]; RP.bf  =(const float*)d_in[42];

  const float* alpha_w=(const float*)d_in[44];
  const float* alpha_b=(const float*)d_in[45];
  const float* color_w=(const float*)d_in[46];
  const float* color_b=(const float*)d_in[47];

  char* ws = (char*)d_ws;
  int*   flat  = (int*)ws;
  float* feats = (float*)(ws + (((size_t)n*4 + 255) & ~(size_t)255));
  float* wvec  = feats + (size_t)n*16;
  float* routs = wvec  + (size_t)n*8;

  int nb = (n+255)/256;
  k_geo  <<<nb, 256, 0, stream>>>(point, timev, GP, flat, n);
  k_table<<<nb, 256, 0, stream>>>(point, timev, flat, TP, feats, wvec, n);
  int nblk = (n+1023)/1024;
  k_render<<<8*nblk, 256, 0, stream>>>(feats, RP, routs, n, nblk);
  k_final<<<nb, 256, 0, stream>>>(routs, wvec, origin,
                                  alpha_w, alpha_b, color_w, color_b,
                                  (float*)d_out, n);
}
